// Attention_54614804136573
// MI455X (gfx1250) — matrix operand order repaired and checked by reading, not compiled, not run on hardware
//
#include <hip/hip_runtime.h>
#include <hip/hip_bf16.h>

#define BB 4
#define NN 1024
#define CC 128
#define PPOS 16      // P*P positions == heads
#define DD 2048
#define HH 16
#define DHH 128
#define MASKNUM 25
#define ATTN_SCALE 0.125f
#define NEG_INF (-1.0e30f)

typedef __attribute__((ext_vector_type(16))) _Float16 v16h;
typedef __attribute__((ext_vector_type(8)))  _Float16 v8h;
typedef __attribute__((ext_vector_type(8)))  float    v8f;
typedef __attribute__((ext_vector_type(4)))  int      v4i;

// gfx1250 async Global->LDS path (ASYNCcnt), guarded: falls back to
// load + ds_store staging if the builtins are absent on this toolchain.
#if defined(__HIP_DEVICE_COMPILE__) && defined(__gfx1250__) && \
    __has_builtin(__builtin_amdgcn_global_load_async_to_lds_b128) && \
    __has_builtin(__builtin_amdgcn_s_wait_asynccnt)
#define HAVE_ASYNC_LDS 1
#else
#define HAVE_ASYNC_LDS 0
#endif

__device__ __forceinline__ void async_copy16(const _Float16* g, _Float16* l) {
#if HAVE_ASYNC_LDS
    __builtin_amdgcn_global_load_async_to_lds_b128(
        (__attribute__((address_space(1))) v4i*)g,
        (__attribute__((address_space(3))) v4i*)l, 0, 0);
#else
    *(v8h*)l = *(const v8h*)g;
#endif
}

__device__ __forceinline__ void wait_async_lds() {
#if HAVE_ASYNC_LDS
    __builtin_amdgcn_s_wait_asynccnt(0);
#endif
}

// ---------------------------------------------------------------------------
// WMMA fragment loaders (f16 16x16x32, wave32; contiguous k order used by this code, same for A and B)
// A (16xK=32): lane<16 -> row=lane, K 0..15; lane>=16 -> row=lane-16,
//              K 16..31.  Two contiguous 16B loads per lane.
// B (K=32x16): lane holds column n=lane&15; K 0..15 (lanes 0-15) or 16..31
//              (lanes 16-31).  One contiguous 32B (2x16B) load from "row n"
//              of the row-major source that represents B transposed.
// ---------------------------------------------------------------------------
__device__ __forceinline__ v16h load_fragA(const _Float16* base, int stride, int lane) {
    int r  = lane & 15;
    int ks = (lane >> 4) << 4;                 // 0 or 16
    const _Float16* p = base + (size_t)r * stride + ks;
    v8h lo = *(const v8h*)(p);
    v8h hi = *(const v8h*)(p + 8);
    return __builtin_shufflevector(lo, hi, 0,1,2,3,4,5,6,7,8,9,10,11,12,13,14,15);
}

__device__ __forceinline__ v16h load_fragB(const _Float16* base, int stride, int lane) {
    int n  = lane & 15;
    int ks = (lane >> 4) << 4;                 // 0 or 16
    const _Float16* p = base + (size_t)n * stride + ks;
    v8h lo = *(const v8h*)(p);
    v8h hi = *(const v8h*)(p + 8);
    return __builtin_shufflevector(lo, hi, 0,1,2,3,4,5,6,7,8,9,10,11,12,13,14,15);
}

__device__ __forceinline__ v8f wmma_f16(v16h a, v16h b, v8f c) {
    return __builtin_amdgcn_wmma_f32_16x16x32_f16(false, a, false, b, (short)0, c, false, false);
}

// ---------------------------------------------------------------------------
// K0: fp32 -> f16 convert
// ---------------------------------------------------------------------------
__global__ __launch_bounds__(256) void cvt_f16_kernel(const float* __restrict__ s,
                                                      _Float16* __restrict__ d, int n) {
    int i = blockIdx.x * 256 + threadIdx.x;
    if (i < n) d[i] = (_Float16)s[i];
}

// ---------------------------------------------------------------------------
// K1: fused QKV projection.  Rows = (b,n,p) = 65536, K = 128, Nout = 128 x3.
// One wave computes a 16x16 tile for q, k, v (A tile reused 3x).
// q also written fp32 for the exact mask path.
// ---------------------------------------------------------------------------
__global__ __launch_bounds__(32) void qkv_wmma_kernel(
    const _Float16* __restrict__ x16, const _Float16* __restrict__ w16,
    const float* __restrict__ bq, const float* __restrict__ bk, const float* __restrict__ bv,
    float* __restrict__ qf, _Float16* __restrict__ q16,
    _Float16* __restrict__ k16, _Float16* __restrict__ v16) {
    int lane = threadIdx.x;
    int rowbase = blockIdx.x * 16;
    int colbase = blockIdx.y * 16;
    const _Float16* ab = x16 + (size_t)rowbase * CC;
    v16h A[4];
#pragma unroll
    for (int c = 0; c < 4; ++c) A[c] = load_fragA(ab + c * 32, CC, lane);
    int col = colbase + (lane & 15);
    int rb  = rowbase + ((lane >> 4) << 3);
#pragma unroll
    for (int pj = 0; pj < 3; ++pj) {
        const _Float16* wb = w16 + (size_t)pj * CC * CC + (size_t)colbase * CC;
        v8f acc = {};
#pragma unroll
        for (int c = 0; c < 4; ++c) {
            v16h Bf = load_fragB(wb + c * 32, CC, lane);   // B = W^T, cols = W rows
            acc = wmma_f16(A[c], Bf, acc);
        }
        const float* bias = (pj == 0) ? bq : (pj == 1) ? bk : bv;
        float bval = bias[col];
        _Float16* o16 = (pj == 0) ? q16 : (pj == 1) ? k16 : v16;
#pragma unroll
        for (int r = 0; r < 8; ++r) {
            float val = acc[r] + bval;
            size_t idx = (size_t)(rb + r) * CC + col;
            if (pj == 0) qf[idx] = val;
            o16[idx] = (_Float16)val;
        }
    }
}

// ---------------------------------------------------------------------------
// K2: token-importance mask.  One block per batch, 1024 threads (1 per token).
// logits -> softmax over N -> exact rank-(MASKNUM-1) threshold -> snap to 1.
// ---------------------------------------------------------------------------
__global__ __launch_bounds__(1024) void token_mask_kernel(
    const float* __restrict__ qf, const float* __restrict__ Wl,
    const float* __restrict__ bl, float* __restrict__ maskb) {
    __shared__ float sv[NN];
    __shared__ float red[NN];
    __shared__ float thr;
    int b = blockIdx.x, n = threadIdx.x;
    const float* qrow = qf + (size_t)(b * NN + n) * DD;
    float acc = 0.f;
    for (int p = 0; p < PPOS; ++p) {
        const float* pr = qrow + p * CC;
#pragma unroll 4
        for (int c = 0; c < CC; ++c) acc += pr[c] * Wl[c];
    }
    float logit = acc * (1.f / 16.f) + bl[0];
    sv[n] = logit;
    red[n] = logit;
    __syncthreads();
    for (int s = NN / 2; s > 0; s >>= 1) {
        if (n < s) red[n] = fmaxf(red[n], red[n + s]);
        __syncthreads();
    }
    float mx = red[0];
    __syncthreads();
    float e = expf(logit - mx);
    red[n] = e;
    __syncthreads();
    for (int s = NN / 2; s > 0; s >>= 1) {
        if (n < s) red[n] += red[n + s];
        __syncthreads();
    }
    float sm = e / red[0];
    // exact rank (ties broken by index -> unique ranks 0..1023)
    int rank = 0;
    for (int j = 0; j < NN; ++j) {
        float vj = sv[j];
        rank += (vj < logit || (vj == logit && j < n)) ? 1 : 0;
    }
    if (rank == MASKNUM - 1) thr = sm;     // MASKNUM-th smallest softmax value
    __syncthreads();
    maskb[b * NN + n] = (sm > thr) ? 1.0f : sm;
}

// ---------------------------------------------------------------------------
// K3: apply mask to k (in place) and build masked V^T (b,h,dh,n) for PV WMMA.
// ---------------------------------------------------------------------------
__global__ __launch_bounds__(256) void mask_scale_kernel(
    _Float16* __restrict__ k16, const _Float16* __restrict__ v16,
    _Float16* __restrict__ vt16, const float* __restrict__ maskb) {
    int tok = blockIdx.x;              // b*NN + n
    int b = tok >> 10, n = tok & 1023;
    float m = maskb[tok];
    for (int d = threadIdx.x; d < DD; d += 256) {
        size_t idx = (size_t)tok * DD + d;
        k16[idx] = (_Float16)((float)k16[idx] * m);
        float vv = (float)v16[idx] * m;
        int h = d >> 7, o = d & 127;
        vt16[(((size_t)(b * HH + h)) * DHH + o) * NN + n] = (_Float16)vv;
    }
}

// ---------------------------------------------------------------------------
// K4: flash-style attention.  One wave handles 16 query rows of one (b,h).
// Scores computed transposed: S^T = K_tile(16x32k) x Q^T  -> softmax is
// in-lane over 8 regs + one shfl_xor(16); exp(S^T) maps directly onto the
// B-operand layout of the PV WMMA (out^T = V^T_tile x P^T).
// ---------------------------------------------------------------------------
__global__ __launch_bounds__(32) void flash_attn_kernel(
    const _Float16* __restrict__ q16, const _Float16* __restrict__ k16,
    const _Float16* __restrict__ vt16, _Float16* __restrict__ att16) {
    int lane = threadIdx.x;
    int qt = blockIdx.x;               // query tile 0..63
    int bh = blockIdx.y;               // b*HH + h
    int b = bh >> 4, h = bh & 15;
    const _Float16* qbase = q16 + (((size_t)(b * NN + qt * 16)) * HH + h) * DHH;
    const _Float16* kbase = k16 + (((size_t)(b * NN)) * HH + h) * DHH;
    const _Float16* vtb   = vt16 + ((size_t)bh * DHH) * NN;

    v16h Qb[4];
#pragma unroll
    for (int c = 0; c < 4; ++c) Qb[c] = load_fragB(qbase + c * 32, DD, lane);

    float m = NEG_INF, l = 0.f;
    v8f acc[8] = {};                   // out^T tiles: d = t*16 + r + 8*(lane>>4), i = lane&15
    bool hiHalf = lane >= 16;

    for (int jb = 0; jb < NN; jb += 32) {
        // prefetch next K block: lane L pulls row jb+32+L's head slice (256B)
        if (jb + 32 < NN) {
            const _Float16* nk = kbase + (size_t)(jb + 32 + lane) * DD;
            __builtin_prefetch(nk, 0, 3);
            __builtin_prefetch(nk + 64, 0, 3);
        }
        v8f s0 = {}, s1 = {};
        const _Float16* ka0 = kbase + (size_t)jb * DD;
        const _Float16* ka1 = ka0 + (size_t)16 * DD;
#pragma unroll
        for (int c = 0; c < 4; ++c) {
            v16h K0 = load_fragA(ka0 + c * 32, DD, lane);
            s0 = wmma_f16(K0, Qb[c], s0);
            v16h K1 = load_fragA(ka1 + c * 32, DD, lane);
            s1 = wmma_f16(K1, Qb[c], s1);
        }
        // online softmax over the 32 new keys (column i = lane&15)
        float tm = NEG_INF;
#pragma unroll
        for (int r = 0; r < 8; ++r) {
            s0[r] *= ATTN_SCALE; s1[r] *= ATTN_SCALE;
            tm = fmaxf(tm, fmaxf(s0[r], s1[r]));
        }
        tm = fmaxf(tm, __shfl_xor(tm, 16, 32));
        float mnew = fmaxf(m, tm);
        float corr = expf(m - mnew);
        float p0[8], p1[8], psum = 0.f;
#pragma unroll
        for (int r = 0; r < 8; ++r) {
            p0[r] = expf(s0[r] - mnew);
            p1[r] = expf(s1[r] - mnew);
            psum += p0[r] + p1[r];
        }
        l = l * corr + psum + __shfl_xor(psum, 16, 32);
        // pack P^T as B operand: lane<16 needs j 0..7 (own s0) + 8..15 (partner s0);
        // lane>=16 needs j 16..23 (partner s1) + 24..31 (own s1)
        v16h P;
#pragma unroll
        for (int r = 0; r < 8; ++r) {
            float o0 = __shfl_xor(p0[r], 16, 32);
            float o1 = __shfl_xor(p1[r], 16, 32);
            P[r]     = (_Float16)(hiHalf ? o1 : p0[r]);
            P[8 + r] = (_Float16)(hiHalf ? p1[r] : o0);
        }
#pragma unroll
        for (int t = 0; t < 8; ++t) {
#pragma unroll
            for (int r = 0; r < 8; ++r) acc[t][r] *= corr;
            v16h Va = load_fragA(vtb + ((size_t)(t * 16)) * NN + jb, NN, lane);
            acc[t] = wmma_f16(Va, P, acc[t]);
        }
        m = mnew;
    }
    float inv = 1.f / l;
    int i = lane & 15;
    int tok = qt * 16 + i;
    _Float16* ob = att16 + ((size_t)(b * NN + tok)) * DD + h * DHH;
    int doff = (lane >> 4) << 3;
#pragma unroll
    for (int t = 0; t < 8; ++t)
#pragma unroll
        for (int r = 0; r < 8; ++r)
            ob[t * 16 + r + doff] = (_Float16)(acc[t][r] * inv);
}

// ---------------------------------------------------------------------------
// K5: output projection  y = attn @ Wo^T + bo.  M=4096, K=2048, N=2048.
// 4 waves / 128 threads per workgroup, 64x64 output tile, A/B chunks staged
// in LDS via double-buffered async Global->LDS copies (ASYNCcnt).
// ---------------------------------------------------------------------------
__global__ __launch_bounds__(128) void out_proj_kernel(
    const _Float16* __restrict__ att16, const _Float16* __restrict__ wo16,
    const float* __restrict__ bo, float* __restrict__ y) {
    __shared__ _Float16 ldsA[2][64 * 32];
    __shared__ _Float16 ldsB[2][64 * 32];
    int tid  = threadIdx.x;
    int lane = tid & 31;
    int wave = tid >> 5;               // 0..3 -> 16-row stripe of the 64-row tile
    int rowbase = blockIdx.x * 64;
    int colbase = blockIdx.y * 64;
    const _Float16* ab = att16 + (size_t)rowbase * DD;
    const _Float16* wb = wo16  + (size_t)colbase * DD;

    auto stage = [&](int buf, int kc) {
        int kbase = kc * 32;
#pragma unroll
        for (int i = 0; i < 2; ++i) {
            int c   = tid + 128 * i;       // 0..255 16B-chunks
            int row = c >> 2;              // 0..63
            int seg = (c & 3) << 3;        // 0,8,16,24 halves
            async_copy16(ab + (size_t)row * DD + kbase + seg, &ldsA[buf][c * 8]);
            async_copy16(wb + (size_t)row * DD + kbase + seg, &ldsB[buf][c * 8]);
        }
    };

    stage(0, 0);
    wait_async_lds();
    __syncthreads();

    v8f acc[4] = {};
    const int KC = DD / 32;               // 64
    for (int kc = 0; kc < KC; ++kc) {
        if (kc + 1 < KC) stage((kc + 1) & 1, kc + 1);   // overlap with compute
        v16h Af = load_fragA(&ldsA[kc & 1][(size_t)(wave * 16) * 32], 32, lane);
#pragma unroll
        for (int ct = 0; ct < 4; ++ct) {
            v16h Bf = load_fragB(&ldsB[kc & 1][(size_t)(ct * 16) * 32], 32, lane);
            acc[ct] = wmma_f16(Af, Bf, acc[ct]);
        }
        if (kc + 1 < KC) {
            wait_async_lds();
            __syncthreads();
        }
    }

    int rb = rowbase + wave * 16 + ((lane >> 4) << 3);
#pragma unroll
    for (int ct = 0; ct < 4; ++ct) {
        int col = colbase + ct * 16 + (lane & 15);
        float bias = bo[col];
#pragma unroll
        for (int r = 0; r < 8; ++r)
            y[(size_t)(rb + r) * DD + col] = acc[ct][r] + bias;
    }
}

// ---------------------------------------------------------------------------
extern "C" void kernel_launch(void* const* d_in, const int* in_sizes, int n_in,
                              void* d_out, int out_size, void* d_ws, size_t ws_size,
                              hipStream_t stream) {
    (void)in_sizes; (void)n_in; (void)out_size; (void)ws_size;
    const float* x  = (const float*)d_in[0];
    const float* Wq = (const float*)d_in[1];
    const float* bq = (const float*)d_in[2];
    const float* Wk = (const float*)d_in[3];
    const float* bk = (const float*)d_in[4];
    const float* Wv = (const float*)d_in[5];
    const float* bv = (const float*)d_in[6];
    const float* Wl = (const float*)d_in[7];
    const float* bl = (const float*)d_in[8];
    const float* Wo = (const float*)d_in[9];
    const float* bo = (const float*)d_in[10];
    float* y = (float*)d_out;

    char* wp = (char*)d_ws;
    auto alloc = [&](size_t bytes) -> void* {
        void* p = (void*)wp;
        wp += (bytes + 255) & ~(size_t)255;
        return p;
    };
    _Float16* x16   = (_Float16*)alloc((size_t)BB * NN * DD * 2);
    _Float16* w16   = (_Float16*)alloc((size_t)3 * CC * CC * 2);
    _Float16* wo16  = (_Float16*)alloc((size_t)DD * DD * 2);
    float*    qf    = (float*)   alloc((size_t)BB * NN * DD * 4);
    _Float16* q16   = (_Float16*)alloc((size_t)BB * NN * DD * 2);
    _Float16* k16   = (_Float16*)alloc((size_t)BB * NN * DD * 2);
    _Float16* v16p  = (_Float16*)alloc((size_t)BB * NN * DD * 2);
    _Float16* vt16  = (_Float16*)alloc((size_t)BB * NN * DD * 2);
    _Float16* att16 = (_Float16*)alloc((size_t)BB * NN * DD * 2);
    float*    maskb = (float*)   alloc((size_t)BB * NN * 4);

    auto cvt = [&](const float* s, _Float16* d, int n) {
        cvt_f16_kernel<<<dim3((n + 255) / 256), dim3(256), 0, stream>>>(s, d, n);
    };
    cvt(x,  x16,  BB * NN * DD);
    cvt(Wq, w16,               CC * CC);
    cvt(Wk, w16 + CC * CC,     CC * CC);
    cvt(Wv, w16 + 2 * CC * CC, CC * CC);
    cvt(Wo, wo16, DD * DD);

    qkv_wmma_kernel<<<dim3(BB * NN * PPOS / 16, CC / 16), dim3(32), 0, stream>>>(
        x16, w16, bq, bk, bv, qf, q16, k16, v16p);
    token_mask_kernel<<<dim3(BB), dim3(NN), 0, stream>>>(qf, Wl, bl, maskb);
    mask_scale_kernel<<<dim3(BB * NN), dim3(256), 0, stream>>>(k16, v16p, vt16, maskb);
    flash_attn_kernel<<<dim3(NN / 16, BB * HH), dim3(32), 0, stream>>>(q16, k16, vt16, att16);
    out_proj_kernel<<<dim3(BB * NN / 64, DD / 64), dim3(128), 0, stream>>>(att16, wo16, bo, y);
}
